// PSBlock_53721450939062
// MI455X (gfx1250) — matrix operand order repaired and checked by reading, not compiled, not run on hardware
//
#include <hip/hip_runtime.h>
#include <math.h>

// ---------------- problem constants ----------------
#define BB   32
#define TT   2016
#define NX   307     // node channels
#define NP   320     // padded channels (mult of 32)
#define FF   1009    // T/2+1 rfft bins
#define FP   1024    // padded freq rows
#define LM   4032    // max padded period length (<= 2T)
#define KTOP 2
#define PMAX 64      // max #periods handled in period attention

typedef unsigned short u16;
typedef __attribute__((ext_vector_type(16))) __bf16 v16bf;
typedef __attribute__((ext_vector_type(8)))  float  v8f;
typedef __attribute__((ext_vector_type(4)))  float  v4f;
typedef __attribute__((ext_vector_type(8)))  u16    v8u16;
typedef __attribute__((ext_vector_type(16))) u16    v16u16;

struct Plan {
  int top[KTOP]; int p[KTOP]; int P[KTOP]; int L[KTOP];
  float pw[BB][KTOP];
};

__device__ __forceinline__ u16 f2bf(float f) {
  unsigned int u = __builtin_bit_cast(unsigned int, f);
  unsigned int r = u + 0x7FFFu + ((u >> 16) & 1u);
  return (u16)(r >> 16);
}
__device__ __forceinline__ float bf2f(u16 h) {
  unsigned int u = ((unsigned int)h) << 16;
  return __builtin_bit_cast(float, u);
}

// =====================================================================
// bf16 WMMA GEMM (always "NT"):  C[m,n] = sum_k A[m,k] * Bt[n,k]
//   A  row-major [M, K]      (contiguous in k)
//   Bt row-major [Ncols, K]  (contiguous in k)  -> both fragments are two
//                                                  16-byte loads per lane
// C f32 or bf16; OUT_T stores C transposed (C[n*ldc+m], one v8u16 / two v4f
// per lane since accumulator rows are contiguous); optional per-m f32 bias.
// Block = 8 waves arranged 2(M) x 4(N); block tile 32x64.
// =====================================================================
template<bool OUT_T, bool OUT_BF16, bool BIAS>
__global__ __launch_bounds__(256)
void gemm_bf16_wmma(const u16* __restrict__ A, int lda, unsigned long long aStride,
                    const u16* __restrict__ Bt, int ldb, unsigned long long bStride,
                    void* __restrict__ Cv, int ldc, unsigned long long cStride,
                    const float* __restrict__ bias,
                    int M, int Ncols, int Kdim)
{
  const int lane = threadIdx.x;             // 0..31 (wave32)
  const int wave = threadIdx.y;             // 0..7 -> 2x4 (M x N)
  const int tm = blockIdx.y * 32 + (wave >> 2) * 16;
  const int tn = blockIdx.x * 64 + (wave & 3) * 16;
  if (tm >= M || tn >= Ncols) return;
  const unsigned long long z = blockIdx.z;
  A  += z * aStride;
  Bt += z * bStride;

  const int lhalf = lane >> 4;              // 0 / 1
  const int l16   = lane & 15;
  // A fragment (ISA 16x32 layout): elems 0..7 = K kk+off.. , 8..15 = K kk+off+16..
  const u16* apRow = A  + (unsigned long long)(tm + l16) * lda + (lhalf ? 8 : 0);
  // B fragment (same as A): elems 0..7 = K kk+off.. , 8..15 = K kk+off+16..
  const u16* bpRow = Bt + (unsigned long long)(tn + l16) * ldb + (lhalf ? 8 : 0);

  v8f acc = {};
  for (int kk = 0; kk < Kdim; kk += 32) {
    const u16* ap = apRow + kk;
    const u16* bp = bpRow + kk;
    __builtin_prefetch(ap + 128, 0, 3);     // -> global_prefetch_b8
    __builtin_prefetch(bp + 128, 0, 3);
    v8u16 alo = *(const v8u16*)(ap);
    v8u16 ahi = *(const v8u16*)(ap + 16);
    v8u16 blo = *(const v8u16*)(bp);
    v8u16 bhi = *(const v8u16*)(bp + 16);
    v16u16 af, bf;
#pragma unroll
    for (int j = 0; j < 8; ++j) {
      af[j] = alo[j]; af[8 + j] = ahi[j];
      bf[j] = blo[j]; bf[8 + j] = bhi[j];
    }
    acc = __builtin_amdgcn_wmma_f32_16x16x32_bf16(
        false, __builtin_bit_cast(v16bf, af),
        false, __builtin_bit_cast(v16bf, bf),
        (short)0, acc, false, false);
  }

  // C layout: VGPR r -> row tm + r + 8*lhalf, col tn + l16
  const int cCol  = tn + l16;
  const int cRow0 = tm + (lhalf ? 8 : 0);
  if constexpr (BIAS) {
#pragma unroll
    for (int r = 0; r < 8; ++r) acc[r] += bias[cRow0 + r];
  }
  if constexpr (OUT_T) {
    // transposed store: contiguous rows per lane -> vector store
    if constexpr (OUT_BF16) {
      v8u16 sv;
#pragma unroll
      for (int r = 0; r < 8; ++r) sv[r] = f2bf(acc[r]);
      *(v8u16*)((u16*)Cv + z * cStride + (unsigned long long)cCol * ldc + cRow0) = sv;
    } else {
      v4f s0, s1;
#pragma unroll
      for (int r = 0; r < 4; ++r) { s0[r] = acc[r]; s1[r] = acc[4 + r]; }
      float* cp = (float*)Cv + z * cStride + (unsigned long long)cCol * ldc + cRow0;
      *(v4f*)(cp)     = s0;
      *(v4f*)(cp + 4) = s1;
    }
  } else {
#pragma unroll
    for (int r = 0; r < 8; ++r) {
      unsigned long long idx = z * cStride + (unsigned long long)(cRow0 + r) * ldc + cCol;
      if constexpr (OUT_BF16) ((u16*)Cv)[idx] = f2bf(acc[r]);
      else                    ((float*)Cv)[idx] = acc[r];
    }
  }
}

// =====================================================================
// Prep kernels
// =====================================================================
__global__ __launch_bounds__(256)
void dft_tables_kernel(u16* __restrict__ wc, u16* __restrict__ ws) {
  unsigned long long i = (unsigned long long)blockIdx.x * 256 + threadIdx.x;
  if (i >= (unsigned long long)FP * TT) return;
  int f = (int)(i / TT), t = (int)(i % TT);
  float c = 0.f, s = 0.f;
  if (f < FF) {
    long long ph = ((long long)f * t) % TT;     // exact phase reduction
    float ang = -6.283185307179586f * (float)ph / (float)TT;
    s = __sinf(ang); c = __cosf(ang);
  }
  wc[i] = f2bf(c); ws[i] = f2bf(s);
}

// x[B,T,N] f32 -> channel-major xcm[B,NP,T] bf16 (zero pad channels)
__global__ __launch_bounds__(256)
void build_xcm_kernel(const float* __restrict__ x, u16* __restrict__ xc) {
  unsigned long long i = (unsigned long long)blockIdx.x * 256 + threadIdx.x;
  if (i >= (unsigned long long)BB * NP * TT) return;
  int t = (int)(i % TT); int n = (int)((i / TT) % NP); int b = (int)(i / ((unsigned long long)NP * TT));
  float v = (n < NX) ? x[((unsigned long long)b * TT + t) * NX + n] : 0.f;
  xc[i] = f2bf(v);
}

// x[B,T,N] -> time-major padded xtm[B,LM,NP] bf16 (zeros for t>=T, n>=NX)
__global__ __launch_bounds__(256)
void build_xtm_kernel(const float* __restrict__ x, u16* __restrict__ xt) {
  unsigned long long i = (unsigned long long)blockIdx.x * 256 + threadIdx.x;
  if (i >= (unsigned long long)BB * LM * NP) return;
  int n = (int)(i % NP); int l = (int)((i / NP) % LM); int b = (int)(i / ((unsigned long long)NP * LM));
  float v = (n < NX && l < TT) ? x[((unsigned long long)b * TT + l) * NX + n] : 0.f;
  xt[i] = f2bf(v);
}

// [NX,NX] f32 -> [NP,NP] bf16 zero-padded
__global__ __launch_bounds__(256)
void wpad_kernel(const float* __restrict__ src, u16* __restrict__ dst) {
  int i = blockIdx.x * 256 + threadIdx.x;
  if (i >= NP * NP) return;
  int r = i / NP, c = i % NP;
  dst[i] = (r < NX && c < NX) ? f2bf(src[r * NX + c]) : (u16)0;
}
__global__ __launch_bounds__(256)
void bpad_kernel(const float* __restrict__ src, float* __restrict__ dst) {
  int i = blockIdx.x * 256 + threadIdx.x;
  if (i >= NP) return;
  dst[i] = (i < NX) ? src[i] : 0.f;
}

// =====================================================================
// Spectrum reduction + top-k plan
// =====================================================================
__global__ __launch_bounds__(256)
void amp_reduce_kernel(const float* __restrict__ Re, const float* __restrict__ Im,
                       float* __restrict__ freq, float* __restrict__ pwall) {
  const int f = blockIdx.x;       // < FF
  const int b = blockIdx.y;
  __shared__ float red[256];
  float s = 0.f;
  for (int n = threadIdx.x; n < NX; n += 256) {
    unsigned long long idx = (((unsigned long long)b * FP) + f) * NP + n;
    float re = Re[idx], im = Im[idx];
    s += sqrtf(re * re + im * im);
  }
  red[threadIdx.x] = s; __syncthreads();
  for (int st = 128; st > 0; st >>= 1) {
    if (threadIdx.x < st) red[threadIdx.x] += red[threadIdx.x + st];
    __syncthreads();
  }
  if (threadIdx.x == 0) {
    float m = red[0] / (float)NX;          // amp.mean over nodes
    pwall[b * FP + f] = m;
    atomicAdd(&freq[f], m / (float)BB);    // mean over batch
  }
}

__global__ void topk_plan_kernel(const float* __restrict__ freq,
                                 const float* __restrict__ pwall,
                                 Plan* __restrict__ plan) {
  __shared__ int stop[KTOP];
  if (threadIdx.x == 0) {
    int i1 = 1; float v1 = -1.f;
    for (int f = 1; f < FF; ++f) { float v = freq[f]; if (v > v1) { v1 = v; i1 = f; } }
    int i2 = (i1 == 1) ? 2 : 1; float v2 = -1.f;
    for (int f = 1; f < FF; ++f) { if (f == i1) continue; float v = freq[f]; if (v > v2) { v2 = v; i2 = f; } }
    int tops[KTOP] = { i1, i2 };
    for (int k = 0; k < KTOP; ++k) {
      int tp = tops[k];
      int p = TT / tp; if (p < 1) p = 1;
      int P = (TT + p - 1) / p;
      plan->top[k] = tp; plan->p[k] = p; plan->P[k] = P; plan->L[k] = P * p;
      stop[k] = tp;
    }
  }
  __syncthreads();
  if (threadIdx.x < BB) {
    int b = threadIdx.x;
    float a = pwall[b * FP + stop[0]];
    float c = pwall[b * FP + stop[1]];
    float mx = fmaxf(a, c);
    float ea = __expf(a - mx), ec = __expf(c - mx);
    float inv = 1.f / (ea + ec);
    plan->pw[b][0] = ea * inv;
    plan->pw[b][1] = ec * inv;
  }
}

// =====================================================================
// Period attention: one block per (b,n). q,k,v rows staged bf16 in LDS,
// S[P,P] f32 in LDS, softmax over period index, O = S x V.
// Time-major buffers: element (b,l,n) at ((b*LM)+l)*NP + n.
// =====================================================================
__global__ __launch_bounds__(256)
void period_attn_kernel(const Plan* __restrict__ plan, int k,
                        const u16* __restrict__ qkv, u16* __restrict__ o) {
  const int n = blockIdx.x, b = blockIdx.y, tid = threadIdx.x;
  const unsigned long long base = (unsigned long long)b * LM * NP + n;
  const unsigned long long span = (unsigned long long)BB * LM * NP;
  if (n >= NX) {                       // zero padded channel cols (read by proj GEMM)
    for (int l = tid; l < TT; l += 256) o[base + (unsigned long long)l * NP] = 0;
    return;
  }
  const int p = plan->p[k];
  int P = plan->P[k]; if (P > PMAX) P = PMAX;
  const int Lc = P * p;

  __shared__ u16 shq[LM + 64], shk[LM + 64], shv[LM + 64];
  __shared__ float S[PMAX * PMAX];

  for (int l = tid; l < Lc; l += 256) {
    unsigned long long g = base + (unsigned long long)l * NP;
    shq[l] = qkv[g];
    shk[l] = qkv[span + g];
    shv[l] = qkv[2 * span + g];
  }
  __syncthreads();

  const float scale = rsqrtf((float)p);
  for (int idx = tid; idx < P * P; idx += 256) {
    int pi = idx / P, qi = idx - pi * P;
    const u16* qr = shq + pi * p;
    const u16* kr = shk + qi * p;
    float s = 0.f;
    for (int f = 0; f < p; ++f) s += bf2f(qr[f]) * bf2f(kr[f]);
    S[pi * PMAX + qi] = s * scale;
  }
  __syncthreads();

  if (tid < P) {
    float mx = -1e30f;
    for (int q = 0; q < P; ++q) mx = fmaxf(mx, S[tid * PMAX + q]);
    float sum = 0.f;
    for (int q = 0; q < P; ++q) { float e = __expf(S[tid * PMAX + q] - mx); S[tid * PMAX + q] = e; sum += e; }
    float inv = 1.f / sum;
    for (int q = 0; q < P; ++q) S[tid * PMAX + q] *= inv;
  }
  __syncthreads();

  for (int l = tid; l < Lc; l += 256) {
    int pi = l / p, f = l - pi * p;
    float a = 0.f;
    for (int q = 0; q < P; ++q) a += S[pi * PMAX + q] * bf2f(shv[q * p + f]);
    o[base + (unsigned long long)l * NP] = f2bf(a);
  }
  for (int l = Lc + tid; l < TT; l += 256)   // only if P was capped
    o[base + (unsigned long long)l * NP] = 0;
}

// =====================================================================
// Spatial softmax: scores[B,NP,NP] f32 -> att[B,NP,NP] bf16 (row softmax m<NX)
// =====================================================================
__global__ __launch_bounds__(256)
void spat_softmax_kernel(const float* __restrict__ sc, u16* __restrict__ att) {
  const int n = blockIdx.x, b = blockIdx.y, tid = threadIdx.x;
  const float* row = sc + (((unsigned long long)b * NP) + n) * NP;
  u16* orow = att + (((unsigned long long)b * NP) + n) * NP;
  if (n >= NX) { for (int m = tid; m < NP; m += 256) orow[m] = 0; return; }
  __shared__ float red[256];
  const float scale = rsqrtf((float)NX);
  float mx = -1e30f;
  for (int m = tid; m < NX; m += 256) mx = fmaxf(mx, row[m] * scale);
  red[tid] = mx; __syncthreads();
  for (int st = 128; st > 0; st >>= 1) { if (tid < st) red[tid] = fmaxf(red[tid], red[tid + st]); __syncthreads(); }
  mx = red[0]; __syncthreads();
  float sum = 0.f;
  for (int m = tid; m < NX; m += 256) sum += __expf(row[m] * scale - mx);
  red[tid] = sum; __syncthreads();
  for (int st = 128; st > 0; st >>= 1) { if (tid < st) red[tid] += red[tid + st]; __syncthreads(); }
  const float inv = 1.f / red[0];
  for (int m = tid; m < NX; m += 256) orow[m] = f2bf(__expf(row[m] * scale - mx) * inv);
  for (int m = NX + tid; m < NP; m += 256) orow[m] = 0;
}

// =====================================================================
// Output init / accumulate (res is time-major [B,T,NP] -> coalesced)
// =====================================================================
__global__ __launch_bounds__(256)
void init_out_kernel(const float* __restrict__ x, float* __restrict__ out) {
  unsigned long long i = (unsigned long long)blockIdx.x * 256 + threadIdx.x;
  if (i >= (unsigned long long)BB * TT * NX) return;
  out[i] = x[i];
}
__global__ __launch_bounds__(256)
void accum_kernel(const Plan* __restrict__ plan, int k,
                  const float* __restrict__ res, float* __restrict__ out) {
  unsigned long long i = (unsigned long long)blockIdx.x * 256 + threadIdx.x;
  if (i >= (unsigned long long)BB * TT * NX) return;
  int n = (int)(i % NX);
  unsigned long long bt = i / NX;                 // b*TT + t
  int b = (int)(bt / TT);
  float w = plan->pw[b][k];
  out[i] += w * res[bt * NP + n];
}

// =====================================================================
// Host-side orchestration
// =====================================================================
static inline size_t al256(size_t x) { return (x + 255) & ~(size_t)255; }

extern "C" void kernel_launch(void* const* d_in, const int* in_sizes, int n_in,
                              void* d_out, int out_size, void* d_ws, size_t ws_size,
                              hipStream_t stream) {
  (void)in_sizes; (void)n_in; (void)out_size; (void)ws_size;
  const float* x        = (const float*)d_in[0];   // [B,T,N]
  const float* pa_qkv   = (const float*)d_in[1];   // [K,3,N,N]
  const float* pa_projw = (const float*)d_in[2];   // [K,N,N]
  const float* pa_projb = (const float*)d_in[3];   // [K,N]
  const float* sa_qkv   = (const float*)d_in[4];   // [K,3,N,N]
  const float* sa_projw = (const float*)d_in[5];   // [K,N,N]
  const float* sa_projb = (const float*)d_in[6];   // [K,N]
  float* out = (float*)d_out;

  // ---- workspace layout ----
  char* ws = (char*)d_ws; size_t cur = 0;
  auto grab = [&](size_t bytes) { size_t o = cur; cur = al256(cur + bytes); return o; };
  Plan*  plan   = (Plan*)(ws + grab(sizeof(Plan)));
  u16*   xcm    = (u16*)(ws + grab((size_t)BB * NP * TT * 2));   // channel-major x
  u16*   xtm    = (u16*)(ws + grab((size_t)BB * LM * NP * 2));   // time-major padded x
  u16*   wcos   = (u16*)(ws + grab((size_t)FP * TT * 2));
  u16*   wsin   = (u16*)(ws + grab((size_t)FP * TT * 2));
  float* Re     = (float*)(ws + grab((size_t)BB * FP * NP * 4));
  float* Im     = (float*)(ws + grab((size_t)BB * FP * NP * 4));
  float* freq   = (float*)(ws + grab((size_t)FP * 4));
  float* pwall  = (float*)(ws + grab((size_t)BB * FP * 4));
  u16*   wts    = (u16*)(ws + grab((size_t)16 * NP * NP * 2));
  float* biasb  = (float*)(ws + grab((size_t)4 * NP * 4));
  u16*   qkvb   = (u16*)(ws + grab((size_t)3 * BB * LM * NP * 2)); // period: time-major; spatial: sq/sk ch-major, sv time-major
  u16*   ob16   = (u16*)(ws + grab((size_t)BB * LM * NP * 2));     // o (time-major); reused as so_tm [B,T,NP]
  u16*   ychb   = (u16*)(ws + grab((size_t)BB * TT * NP * 2));     // time-major
  float* scores = (float*)(ws + grab((size_t)BB * NP * NP * 4));
  u16*   attb   = (u16*)(ws + grab((size_t)BB * NP * NP * 2));
  float* res32  = (float*)(ws + grab((size_t)BB * TT * NP * 4));   // time-major

  auto wslot = [&](int i) { return wts + (size_t)i * NP * NP; };
  // slots: pa_qkv(k,j)=k*3+j, pa_proj(k)=6+k, sa_qkv(k,j)=8+k*3+j, sa_proj(k)=14+k

  const dim3 gblk(32, 8);  // 8 waves (2x4), 16x16 WMMA tile each -> 32x64 block tile

  // ---- 1. prep ----
  {
    unsigned long long tot = (unsigned long long)FP * TT;
    dft_tables_kernel<<<(unsigned)((tot + 255) / 256), 256, 0, stream>>>(wcos, wsin);
  }
  {
    unsigned long long tot = (unsigned long long)BB * NP * TT;
    build_xcm_kernel<<<(unsigned)((tot + 255) / 256), 256, 0, stream>>>(x, xcm);
  }
  {
    unsigned long long tot = (unsigned long long)BB * LM * NP;
    build_xtm_kernel<<<(unsigned)((tot + 255) / 256), 256, 0, stream>>>(x, xtm);
  }
  const int WG = (NP * NP + 255) / 256;
  for (int k = 0; k < KTOP; ++k) {
    for (int j = 0; j < 3; ++j) {
      wpad_kernel<<<WG, 256, 0, stream>>>(pa_qkv + ((size_t)(k * 3 + j)) * NX * NX, wslot(k * 3 + j));
      wpad_kernel<<<WG, 256, 0, stream>>>(sa_qkv + ((size_t)(k * 3 + j)) * NX * NX, wslot(8 + k * 3 + j));
    }
    wpad_kernel<<<WG, 256, 0, stream>>>(pa_projw + (size_t)k * NX * NX, wslot(6 + k));
    wpad_kernel<<<WG, 256, 0, stream>>>(sa_projw + (size_t)k * NX * NX, wslot(14 + k));
    bpad_kernel<<<2, 256, 0, stream>>>(pa_projb + (size_t)k * NX, biasb + (size_t)k * NP);
    bpad_kernel<<<2, 256, 0, stream>>>(sa_projb + (size_t)k * NX, biasb + (size_t)(2 + k) * NP);
  }
  hipMemsetAsync(freq, 0, FP * sizeof(float), stream);

  // ---- 2. DFT: C[f,n] = sum_t W[f,t] * xcm[b][n][t]  (both operands contiguous in t) ----
  {
    dim3 grid((NP + 63) / 64, FP / 32, BB);
    gemm_bf16_wmma<false, false, false><<<grid, gblk, 0, stream>>>(
        wcos, TT, 0ull, xcm, TT, (unsigned long long)NP * TT,
        Re, NP, (unsigned long long)FP * NP, nullptr, FP, NP, TT);
    gemm_bf16_wmma<false, false, false><<<grid, gblk, 0, stream>>>(
        wsin, TT, 0ull, xcm, TT, (unsigned long long)NP * TT,
        Im, NP, (unsigned long long)FP * NP, nullptr, FP, NP, TT);
  }

  // ---- 3. spectrum means + top-k plan + output init ----
  {
    dim3 grid(FF, BB);
    amp_reduce_kernel<<<grid, 256, 0, stream>>>(Re, Im, freq, pwall);
  }
  topk_plan_kernel<<<1, 64, 0, stream>>>(freq, pwall, plan);
  {
    unsigned long long tot = (unsigned long long)BB * TT * NX;
    init_out_kernel<<<(unsigned)((tot + 255) / 256), 256, 0, stream>>>(x, out);
  }

  const unsigned long long spanQKV = (unsigned long long)BB * LM * NP;

  // ---- 4. per-scale chain ----
  for (int k = 0; k < KTOP; ++k) {
    // (a) period qkv: C[n,l] = sum_m W[n,m] xtm[b][l][m]; store transposed -> time-major
    {
      dim3 grid(LM / 64, NP / 32, BB);
      for (int j = 0; j < 3; ++j)
        gemm_bf16_wmma<true, true, false><<<grid, gblk, 0, stream>>>(
            wslot(k * 3 + j), NP, 0ull, xtm, NP, (unsigned long long)LM * NP,
            qkvb + (unsigned long long)j * spanQKV, NP, (unsigned long long)LM * NP,
            nullptr, NP, LM, NP);
    }
    // (b) period attention (VALU; P small)
    {
      dim3 grid(NP, BB);
      period_attn_kernel<<<grid, 256, 0, stream>>>(plan, k, qkvb, ob16);
    }
    // (c) period proj (+bias) -> ych time-major [B,T,NP]
    {
      dim3 grid((TT + 63) / 64, NP / 32, BB);
      gemm_bf16_wmma<true, true, true><<<grid, gblk, 0, stream>>>(
          wslot(6 + k), NP, 0ull, ob16, NP, (unsigned long long)LM * NP,
          ychb, NP, (unsigned long long)TT * NP, biasb + (size_t)k * NP, NP, TT, NP);
    }
    // (d) spatial qkv: sq/sk channel-major [B,NP,T] (normal store), sv time-major
    {
      dim3 grid((TT + 63) / 64, NP / 32, BB);
      gemm_bf16_wmma<false, true, false><<<grid, gblk, 0, stream>>>(
          wslot(8 + k * 3 + 0), NP, 0ull, ychb, NP, (unsigned long long)TT * NP,
          qkvb, TT, (unsigned long long)NP * TT, nullptr, NP, TT, NP);
      gemm_bf16_wmma<false, true, false><<<grid, gblk, 0, stream>>>(
          wslot(8 + k * 3 + 1), NP, 0ull, ychb, NP, (unsigned long long)TT * NP,
          qkvb + spanQKV, TT, (unsigned long long)NP * TT, nullptr, NP, TT, NP);
      gemm_bf16_wmma<true, true, false><<<grid, gblk, 0, stream>>>(
          wslot(8 + k * 3 + 2), NP, 0ull, ychb, NP, (unsigned long long)TT * NP,
          qkvb + 2ull * spanQKV, NP, (unsigned long long)TT * NP, nullptr, NP, TT, NP);
    }
    // (e) scores[n,m] = sum_t sq[n,t] sk[m,t] (both channel-major, contiguous in t)
    {
      dim3 grid(NP / 64, NP / 32, BB);
      gemm_bf16_wmma<false, false, false><<<grid, gblk, 0, stream>>>(
          qkvb, TT, (unsigned long long)NP * TT,
          qkvb + spanQKV, TT, (unsigned long long)NP * TT,
          scores, NP, (unsigned long long)NP * NP, nullptr, NP, NP, TT);
    }
    // (f) row softmax -> att bf16 [NP,NP] (contiguous in m)
    {
      dim3 grid(NP, BB);
      spat_softmax_kernel<<<grid, 256, 0, stream>>>(scores, attb);
    }
    // (g) so[n,t] = sum_m att[n,m] sv_tm[b][t][m]; store transposed -> so time-major
    {
      dim3 grid((TT + 63) / 64, NP / 32, BB);
      gemm_bf16_wmma<true, true, false><<<grid, gblk, 0, stream>>>(
          attb, NP, (unsigned long long)NP * NP,
          qkvb + 2ull * spanQKV, NP, (unsigned long long)TT * NP,
          ob16, NP, (unsigned long long)TT * NP, nullptr, NP, TT, NP);
    }
    // (h) spatial proj (+bias) -> res32 time-major [B,T,NP] f32
    {
      dim3 grid((TT + 63) / 64, NP / 32, BB);
      gemm_bf16_wmma<true, false, true><<<grid, gblk, 0, stream>>>(
          wslot(14 + k), NP, 0ull, ob16, NP, (unsigned long long)TT * NP,
          res32, NP, (unsigned long long)TT * NP, biasb + (size_t)(2 + k) * NP, NP, TT, NP);
    }
    // (i) out += pw[b,k] * res (coalesced)
    {
      unsigned long long tot = (unsigned long long)BB * TT * NX;
      accum_kernel<<<(unsigned)((tot + 255) / 256), 256, 0, stream>>>(plan, k, res32, out);
    }
  }
}
